// CSOCKernelAttention_91207925498224
// MI455X (gfx1250) — matrix operand order repaired and checked by reading, not compiled, not run on hardware
//
#include <hip/hip_runtime.h>
#include <hip/hip_bf16.h>

// ---------------------------------------------------------------------------
// CDNA5 (gfx1250) attention with power-law/exponential decay kernel.
// All matmuls run through v_wmma_f32_16x16x32_bf16 (wave32 WMMA).
// Each K-step: load phase (A + 4 B fragments, distinct registers, one clause)
// then compute phase (4 WMMAs) -> partial s_wait_loadcnt overlap, no spills.
// ---------------------------------------------------------------------------

#define BB   4
#define NN   2048
#define DD   1024
#define MTOT (BB * NN)          // 8192 rows for the QKV projections

typedef __attribute__((ext_vector_type(16))) __bf16 v16bf;
typedef __attribute__((ext_vector_type(8)))  float  v8f;

union FragU {
  v16bf v;
  uint4 q[2];
};

// A fragment: 16x32 (MxK), row-major source, src already offset to (row0, k0).
// Lane m (m<16) holds K = {0..7, 16..23}; lane m+16 holds K = {8..15, 24..31}.
__device__ __forceinline__ v16bf load_frag_a(const __bf16* src, int ld, int lane) {
  const int m  = lane & 15;
  const int kb = (lane < 16) ? 0 : 8;
  const __bf16* p = src + m * ld + kb;
  FragU f;
  f.q[0] = *(const uint4*)(p);        // 8 bf16 = 16B, aligned
  f.q[1] = *(const uint4*)(p + 16);   // next 8 bf16 at K+16
  return f.v;
}

// B fragment: 32x16 (KxN) where B[k,n] = src[n*ld + k] (NT layout: src rows
// are the N dimension, contiguous over K). Lane n (n<16) holds K = {0..7, 16..23},
// lane n+16 holds K = {8..15, 24..31}. src already offset to (n0 row, k0).
__device__ __forceinline__ v16bf load_frag_b(const __bf16* src, int ld, int lane) {
  const int n  = lane & 15;
  const int kb = (lane < 16) ? 0 : 8;
  const __bf16* p = src + n * ld + kb;
  FragU f;
  f.q[0] = *(const uint4*)(p);        // 8 bf16 = 16B, aligned
  f.q[1] = *(const uint4*)(p + 16);   // next 8 bf16 at K+16
  return f.v;
}

__device__ __forceinline__ v8f wmma_bf16(v16bf a, v16bf b, v8f c) {
  return __builtin_amdgcn_wmma_f32_16x16x32_bf16(
      /*neg_a=*/false, a, /*neg_b=*/false, b,
      /*c_mod=*/(short)0, c, /*reuse_a=*/false, /*reuse_b=*/false);
}

// ---------------------------------------------------------------------------
// f32 -> bf16 cast
// ---------------------------------------------------------------------------
__global__ void cast_f32_bf16(const float* __restrict__ src,
                              __bf16* __restrict__ dst, int n) {
  int idx = blockIdx.x * blockDim.x + threadIdx.x;
  if (idx < n) dst[idx] = (__bf16)src[idx];
}

// ---------------------------------------------------------------------------
// Distance-decay kernel K[i,j] = (|i-j|+eps)^(-|alpha|) * exp(-|i-j|/|lam|)
// ---------------------------------------------------------------------------
__global__ void decay_kernel(float* __restrict__ Km,
                             const float* __restrict__ alpha_p,
                             const float* __restrict__ lam_p) {
  int idx = blockIdx.x * blockDim.x + threadIdx.x;
  if (idx >= NN * NN) return;
  const int i = idx >> 11;
  const int j = idx & (NN - 1);
  const float r = fabsf((float)(i - j));
  const float a = fabsf(alpha_p[0]);
  const float l = fabsf(lam_p[0]);
  Km[idx] = powf(r + 1e-4f, -a) * __expf(-r / l);
}

// ---------------------------------------------------------------------------
// Projection GEMM: Y[m,n] = sum_k X[m,k] * W[n,k] + bias[n]   (torch Linear)
// Wave tile: 16 (M) x 64 (N) -> 4 accumulators, A fragment reused 4x.
// Block: 8 waves -> 128 x 64 tile.
// transposed==0: Y row-major bf16 [MTOT, DD]
// transposed==1: Y = vT layout [b][n][m] bf16 (for the PV matmul)
// ---------------------------------------------------------------------------
__global__ void proj_gemm(const __bf16* __restrict__ X,
                          const __bf16* __restrict__ W,
                          const float* __restrict__ bias,
                          __bf16* __restrict__ Y, int transposed) {
  const int tid  = threadIdx.x;
  const int lane = tid & 31;
  const int wv   = tid >> 5;
  const int m0   = blockIdx.y * 128 + wv * 16;
  const int n0   = blockIdx.x * 64;

  const __bf16* Xa = X + (size_t)m0 * DD;
  const __bf16* Wb[4];
#pragma unroll
  for (int t = 0; t < 4; ++t) Wb[t] = W + (size_t)(n0 + 16 * t) * DD;

  v8f acc[4] = {};
#pragma unroll 2
  for (int kk = 0; kk < DD; kk += 32) {
    // load phase: A + all 4 B fragments in distinct registers (one clause)
    v16bf a = load_frag_a(Xa + kk, DD, lane);
    v16bf bf[4];
#pragma unroll
    for (int t = 0; t < 4; ++t) bf[t] = load_frag_b(Wb[t] + kk, DD, lane);
    // compute phase: partial waits overlap WMMA with in-flight loads
#pragma unroll
    for (int t = 0; t < 4; ++t) acc[t] = wmma_bf16(a, bf[t], acc[t]);
  }

  const int l15 = lane & 15;
  const int h8  = (lane < 16) ? 0 : 8;
  for (int t = 0; t < 4; ++t) {
    const int n = n0 + 16 * t + l15;
    const float bv_ = bias[n];
#pragma unroll
    for (int r = 0; r < 8; ++r) {
      const int m = m0 + r + h8;
      const float y = acc[t][r] + bv_;
      if (!transposed) {
        Y[(size_t)m * DD + n] = (__bf16)y;
      } else {
        const int b  = m >> 11;          // m / NN
        const int mm = m & (NN - 1);
        Y[(size_t)b * (DD * NN) + (size_t)n * NN + mm] = (__bf16)y;
      }
    }
  }
}

// ---------------------------------------------------------------------------
// Attention: one workgroup = one (batch, 16-row) tile.
// LDS: q tile 32KB + scores f32 128KB + P bf16 64KB = 224KB (CDNA5 320KB WGP).
// ---------------------------------------------------------------------------
__global__ void attn_kernel(const __bf16* __restrict__ qb,
                            const __bf16* __restrict__ kb,
                            const __bf16* __restrict__ vtb,
                            const float* __restrict__ Km,
                            float* __restrict__ out,
                            float* __restrict__ attn) {
  extern __shared__ char smem[];
  __bf16* qs = (__bf16*)smem;                              // 16*1024 bf16
  float*  S  = (float*)(smem + 16 * DD * 2);               // 16*2048 f32
  __bf16* Ps = (__bf16*)(smem + 16 * DD * 2 + 16 * NN * 4);// 16*2048 bf16

  __shared__ float red[16][16];
  __shared__ float rowmax[16];
  __shared__ float rowsum[16];

  const int tid  = threadIdx.x;
  const int lane = tid & 31;
  const int wv   = tid >> 5;
  const int b    = blockIdx.x >> 7;            // 128 row-tiles per batch
  const int i0   = (blockIdx.x & 127) * 16;
  const int l15  = lane & 15;
  const int h8   = (lane < 16) ? 0 : 8;

  // ---- stage Q tile to LDS --------------------------------------------
  {
    const unsigned int* src =
        (const unsigned int*)(qb + (size_t)(b * NN + i0) * DD);
    unsigned int* dst = (unsigned int*)qs;
    for (int idx = tid; idx < 16 * DD / 2; idx += 256) dst[idx] = src[idx];
  }
  __syncthreads();

  // ---- logits: S[i,j] = (q_i . k_j) / sqrt(D) * Km[i,j] ----------------
  // 32 groups of 4 j-tiles; wave handles groups wv, wv+8, ... (A reuse 4x).
  const float scale = 0.03125f;  // 1/sqrt(1024)
  for (int gi = wv; gi < 32; gi += 8) {
    const int j0 = gi * 64;
    const __bf16* kbase[4];
#pragma unroll
    for (int t = 0; t < 4; ++t)
      kbase[t] = kb + (size_t)(b * NN + j0 + 16 * t) * DD;

    v8f acc[4] = {};
#pragma unroll 2
    for (int kk = 0; kk < DD; kk += 32) {
      v16bf a = load_frag_a(qs + kk, DD, lane);
      v16bf bf[4];
#pragma unroll
      for (int t = 0; t < 4; ++t) bf[t] = load_frag_b(kbase[t] + kk, DD, lane);
#pragma unroll
      for (int t = 0; t < 4; ++t) acc[t] = wmma_bf16(a, bf[t], acc[t]);
    }

#pragma unroll
    for (int t = 0; t < 4; ++t) {
      const int j = j0 + 16 * t + l15;
#pragma unroll
      for (int r = 0; r < 8; ++r) {
        const int i = r + h8;
        S[i * NN + j] = acc[t][r] * scale * Km[(size_t)(i0 + i) * NN + j];
      }
    }
  }
  __syncthreads();

  // ---- softmax over each of the 16 rows (16 threads per row) -----------
  {
    const int row = tid >> 4;
    const int sl  = tid & 15;
    float m = -3.4e38f;
    for (int j = sl; j < NN; j += 16) m = fmaxf(m, S[row * NN + j]);
    red[row][sl] = m;
    __syncthreads();
    if (sl == 0) {
      float mm = red[row][0];
      for (int t = 1; t < 16; ++t) mm = fmaxf(mm, red[row][t]);
      rowmax[row] = mm;
    }
    __syncthreads();
    const float mm = rowmax[row];
    float s = 0.f;
    for (int j = sl; j < NN; j += 16) {
      const float e = __expf(S[row * NN + j] - mm);
      S[row * NN + j] = e;
      s += e;
    }
    red[row][sl] = s;
    __syncthreads();
    if (sl == 0) {
      float ss = 0.f;
      for (int t = 0; t < 16; ++t) ss += red[row][t];
      rowsum[row] = ss;
    }
    __syncthreads();
    const float inv = 1.f / rowsum[row];
    float* arow = attn + (size_t)(b * NN + i0 + row) * NN;
    for (int j = sl; j < NN; j += 16) {
      const float p = S[row * NN + j] * inv;
      arow[j] = p;                 // attn output (f32, mandatory)
      Ps[row * NN + j] = (__bf16)p;
    }
  }
  __syncthreads();

  // ---- out = P @ V : wave wv covers columns [wv*128, wv*128+128), in two
  // sequential 64-column passes (4 accumulators each -> no VGPR spills).
  for (int h = 0; h < 2; ++h) {
    const int nbase = wv * 128 + h * 64;
    const __bf16* vbase[4];
#pragma unroll
    for (int t = 0; t < 4; ++t)
      vbase[t] = vtb + (size_t)(b * DD + nbase + 16 * t) * NN;

    v8f acc[4] = {};
#pragma unroll 2
    for (int kk = 0; kk < NN; kk += 32) {
      v16bf a = load_frag_a(Ps + kk, NN, lane);
      v16bf bf[4];
#pragma unroll
      for (int t = 0; t < 4; ++t) bf[t] = load_frag_b(vbase[t] + kk, NN, lane);
#pragma unroll
      for (int t = 0; t < 4; ++t) acc[t] = wmma_bf16(a, bf[t], acc[t]);
    }

    for (int t = 0; t < 4; ++t) {
      const int n = nbase + 16 * t + l15;
#pragma unroll
      for (int r = 0; r < 8; ++r) {
        const int i = r + h8;
        out[(size_t)(b * NN + i0 + i) * DD + n] = acc[t][r];
      }
    }
  }
}

// ---------------------------------------------------------------------------
// Host launch
// ---------------------------------------------------------------------------
extern "C" void kernel_launch(void* const* d_in, const int* in_sizes, int n_in,
                              void* d_out, int out_size, void* d_ws, size_t ws_size,
                              hipStream_t stream) {
  (void)in_sizes; (void)n_in; (void)out_size; (void)ws_size;

  const float* x     = (const float*)d_in[0];
  const float* Wq    = (const float*)d_in[1];
  const float* bq    = (const float*)d_in[2];
  const float* Wk    = (const float*)d_in[3];
  const float* bk    = (const float*)d_in[4];
  const float* Wv    = (const float*)d_in[5];
  const float* bv    = (const float*)d_in[6];
  const float* alpha = (const float*)d_in[7];
  const float* lam   = (const float*)d_in[8];

  float* out  = (float*)d_out;                       // [B,N,D]
  float* attn = out + (size_t)BB * NN * DD;          // [B,N,N]
  float* Km   = attn + (size_t)BB * NN * NN;         // [N,N]

  char* ws = (char*)d_ws;
  __bf16* xb  = (__bf16*)(ws);                                  // 16.78 MB
  __bf16* Wqb = (__bf16*)(ws + (size_t)16777216);               // 2.1 MB
  __bf16* Wkb = (__bf16*)(ws + (size_t)18874368);
  __bf16* Wvb = (__bf16*)(ws + (size_t)20971520);
  __bf16* qb  = (__bf16*)(ws + (size_t)23068672);               // 16.78 MB
  __bf16* kb  = (__bf16*)(ws + (size_t)39845888);
  __bf16* vtb = (__bf16*)(ws + (size_t)56623104);               // vT [b][n][m]

  const int nx = MTOT * DD;      // 8,388,608
  const int nw = DD * DD;        // 1,048,576

  cast_f32_bf16<<<(nx + 255) / 256, 256, 0, stream>>>(x,  xb,  nx);
  cast_f32_bf16<<<(nw + 255) / 256, 256, 0, stream>>>(Wq, Wqb, nw);
  cast_f32_bf16<<<(nw + 255) / 256, 256, 0, stream>>>(Wk, Wkb, nw);
  cast_f32_bf16<<<(nw + 255) / 256, 256, 0, stream>>>(Wv, Wvb, nw);

  decay_kernel<<<(NN * NN + 255) / 256, 256, 0, stream>>>(Km, alpha, lam);

  dim3 ggrid(DD / 64, MTOT / 128);   // (16, 64)
  proj_gemm<<<ggrid, 256, 0, stream>>>(xb, Wqb, bq, qb,  0);
  proj_gemm<<<ggrid, 256, 0, stream>>>(xb, Wkb, bk, kb,  0);
  proj_gemm<<<ggrid, 256, 0, stream>>>(xb, Wvb, bv, vtb, 1);

  const size_t attn_smem = (size_t)16 * DD * 2      // q tile bf16
                         + (size_t)16 * NN * 4      // scores f32
                         + (size_t)16 * NN * 2;     // P bf16   = 229376 B
  attn_kernel<<<BB * (NN / 16), 256, attn_smem, stream>>>(qb, kb, vtb, Km, out, attn);
}
